// NIII_89404039234166
// MI455X (gfx1250) — matrix operand order repaired and checked by reading, not compiled, not run on hardware
//
#include <hip/hip_runtime.h>

typedef unsigned short u16;
typedef unsigned int   u32;
typedef __attribute__((ext_vector_type(8)))  float  v8f;
typedef __attribute__((ext_vector_type(16))) __bf16 v16bf;
typedef int v4i __attribute__((vector_size(16)));   // matches builtin's V4i

#define AS1 __attribute__((address_space(1)))
#define AS3 __attribute__((address_space(3)))

#if __has_builtin(__builtin_amdgcn_global_load_async_to_lds_b128)
#define ASYNC_LDS 1
#endif

struct alignas(16) U128 { u32 x, y, z, w; };
union Frag { U128 u[2]; v16bf v; };

__device__ __forceinline__ u16 f2bf(float f) {
  union { float f; u32 u; } c; c.f = f;
  u32 u = c.u;
  if ((u & 0x7F800000u) == 0x7F800000u) return (u16)(u >> 16); // inf/nan passthrough
  return (u16)((u + 0x7FFFu + ((u >> 16) & 1u)) >> 16);        // round-nearest-even
}
__device__ __forceinline__ float bf2f(u16 h) {
  union { u32 u; float f; } c; c.u = ((u32)h) << 16; return c.f;
}

__device__ __forceinline__ v8f bwmma(const Frag& a, const Frag& b, v8f c) {
  return __builtin_amdgcn_wmma_f32_16x16x32_bf16(false, a.v, false, b.v,
                                                 (short)0, c, false, false);
}

__device__ __forceinline__ void async_wait0() {
#if defined(ASYNC_LDS)
#if __has_builtin(__builtin_amdgcn_s_wait_asynccnt)
  __builtin_amdgcn_s_wait_asynccnt(0);
#else
  asm volatile("s_wait_asynccnt 0" ::: "memory");
#endif
#endif
}

// ---------------------------------------------------------------------------
// Weight packing: fp32 (N,K[,3,3]) -> bf16 fragment layout [t][kc][nt][lane][16]
// B-fragment (this code's contiguous k order, same as the A loads): lane l -> n = nt*16 + (l&15), k = kc*32 + (l>>4)*16 + j.
// ---------------------------------------------------------------------------
__global__ void k_pack(const float* __restrict__ w, u16* __restrict__ dst,
                       int K, int N, int KC, int NT, int taps, long total) {
  long idx = (long)blockIdx.x * blockDim.x + threadIdx.x;
  if (idx >= total) return;
  int  j    = (int)(idx & 15);
  int  lane = (int)((idx >> 4) & 31);
  long rest = idx >> 9;
  int  nt = (int)(rest % NT); rest /= NT;
  int  kc = (int)(rest % KC);
  int  t  = (int)(rest / KC);
  int n = nt * 16 + (lane & 15);
  int k = kc * 32 + ((lane >> 4) << 4) + j;
  float v = 0.f;
  if (k < K && n < N) v = w[((long)n * K + k) * taps + t];
  dst[idx] = f2bf(v);
}

__global__ void k_cvt(const float* __restrict__ src, u16* __restrict__ dst, int n) {
  int i = blockIdx.x * blockDim.x + threadIdx.x;
  if (i < n) dst[i] = f2bf(src[i]);
}

__global__ void k_copy(const u16* __restrict__ src, int sstride,
                       u16* __restrict__ dst, int dstride, int rows, int cols) {
  long i = (long)blockIdx.x * blockDim.x + threadIdx.x;
  if (i >= (long)rows * cols) return;
  int r = (int)(i / cols), c = (int)(i % cols);
  dst[(long)r * dstride + c] = src[(long)r * sstride + c];
}

// ---------------------------------------------------------------------------
// 3x3 conv (pad 1) as 9-tap implicit GEMM.
// Block = 256 thr (8 waves), 32-pixel strip x 128 output channels.
// Wave = 16px x 32ch (2 accums, A-fragment reused across the n-tile pair).
// Input halo (3 x 34 pixels) staged 32 channels at a time, double-buffered.
// ---------------------------------------------------------------------------
__global__ void __launch_bounds__(256)
k_conv3x3(const u16* __restrict__ in, int in_stride, int Cin, int CinPad,
          const u16* __restrict__ wp, const float* __restrict__ bias,
          u16* __restrict__ out, int out_stride,
          const u16* __restrict__ res, int res_stride,
          int H, int W, int relu) {
  __shared__ u16 lds[2 * 102 * 32];          // 2 buffers: 3x34 pixels x 32 ch
  const int tid = threadIdx.x;
  const int p0 = blockIdx.x * 32;
  const int y = p0 / W, x0 = p0 % W;
  const int KC = CinPad >> 5;

  auto stage = [&](int kc, int bufi) {
    u16* buf = lds + bufi * (102 * 32);
    if (Cin == CinPad) {                     // vectorized: 4 x 16B per pixel
      for (int i = tid; i < 102 * 4; i += 256) {
        int v = i & 3, pix = i >> 2;
        int r = pix / 34, cc = pix % 34;
        int gy = y - 1 + r, gx = x0 - 1 + cc;
        U128 val = {0, 0, 0, 0};
        if (gy >= 0 && gy < H && gx >= 0 && gx < W)
          val = *((const U128*)(in + ((long)gy * W + gx) * in_stride + kc * 32) + v);
        ((U128*)(buf + pix * 32))[v] = val;
      }
    } else {                                 // scalar path (sfe1: Cin=1)
      for (int i = tid; i < 102 * 32; i += 256) {
        int c = i & 31, pix = i >> 5;
        int r = pix / 34, cc = pix % 34;
        int gy = y - 1 + r, gx = x0 - 1 + cc;
        int ch = kc * 32 + c;
        u16 val = 0;
        if (gy >= 0 && gy < H && gx >= 0 && gx < W && ch < Cin)
          val = in[((long)gy * W + gx) * in_stride + ch];
        buf[pix * 32 + c] = val;
      }
    }
  };

  stage(0, 0);
  __syncthreads();

  const int wave = tid >> 5, lane = tid & 31;
  const int mg = wave & 1;                   // pixel half (2 x 16 px)
  const int nt0 = (wave >> 1) * 2;           // n-tiles nt0, nt0+1 (NT == 8)
  const int m = lane & 15, half = lane >> 4;
  const int lp = mg * 16 + m;                // local pixel row in strip

  v8f acc0 = {}, acc1 = {};
  for (int kc = 0; kc < KC; ++kc) {
    const u16* buf = lds + (kc & 1) * (102 * 32);
    if (kc + 1 < KC) stage(kc + 1, (kc + 1) & 1);
    for (int t = 0; t < 9; ++t) {
      const u16* lrow = buf + ((t / 3) * 34 + lp + (t % 3)) * 32;
      const u16* brow = wp + (((long)t * KC + kc) * 8 + nt0) * 512 + lane * 16;
      Frag a, b0, b1;
      a.u[0]  = *(const U128*)(lrow + half * 16);       // k = kb..kb+7
      a.u[1]  = *(const U128*)(lrow + half * 16 + 8);   // k = kb+8..kb+15
      b0.u[0] = *(const U128*)(brow);
      b0.u[1] = *(const U128*)(brow + 8);
      b1.u[0] = *(const U128*)(brow + 512);
      b1.u[1] = *(const U128*)(brow + 520);
      acc0 = bwmma(a, b0, acc0);
      acc1 = bwmma(a, b1, acc1);
    }
    __syncthreads();
  }

  const int n = lane & 15, mh = lane >> 4;
  for (int tile = 0; tile < 2; ++tile) {
    const v8f& acc = tile ? acc1 : acc0;
    const int ng = (nt0 + tile) * 16 + n;
    float bb = bias ? bias[ng] : 0.f;
    for (int r = 0; r < 8; ++r) {
      long p = p0 + mg * 16 + mh * 8 + r;
      float v = acc[r] + bb;
      if (res) v += bf2f(res[p * (long)res_stride + ng]);
      if (relu) v = fmaxf(v, 0.f);
      out[p * (long)out_stride + ng] = f2bf(v);
    }
  }
}

// ---------------------------------------------------------------------------
// GEMM: out[M][out_stride] = act(A[M][Kpad] x Wp + bias [+ res]); bf16 in/out.
// Block = 256 thr (8 waves) -> 64M x 128N tile; wave = 32x32 (2x2 accums).
// A K-slices (64 rows x 32 k) double-buffered in LDS, staged with
// GLOBAL_LOAD_ASYNC_TO_LDS_B128 when the toolchain exposes it.
// ---------------------------------------------------------------------------
__global__ void __launch_bounds__(256)
k_gemm(const u16* __restrict__ A, int Kpad,
       const u16* __restrict__ wp, const float* __restrict__ bias, int Nact,
       u16* __restrict__ out, int out_stride,
       const u16* __restrict__ res, int res_stride,
       int NT, int relu) {
  __shared__ u16 lds[2 * 2048];             // 2 buffers of 64 rows x 32 k
  const int tid = threadIdx.x;
  const long p0 = (long)blockIdx.x * 64;
  const int KC = Kpad >> 5;

  auto stage = [&](int kc, int buf) {       // 64 rows x 4 U128 = 256 transfers
    int r = tid >> 2, v = tid & 3;
    const U128* s = (const U128*)(A + (p0 + r) * Kpad + kc * 32) + v;
#if defined(ASYNC_LDS)
    AS1 v4i* g1 = (AS1 v4i*)(v4i*)(U128*)s;                    // drop const, AS1
    AS3 v4i* l3 = (AS3 v4i*)((AS3 u16*)lds + buf * 2048 + tid * 8);
    __builtin_amdgcn_global_load_async_to_lds_b128(g1, l3, 0, 0);
#else
    ((U128*)(lds + buf * 2048))[tid] = *s;
#endif
  };

  stage(0, 0);
  async_wait0();
  __syncthreads();

  const int wave = tid >> 5, lane = tid & 31;
  const int mg = wave & 1;                  // 2 m-groups of 32 rows
  const int ng = wave >> 1;                 // 4 n-groups of 32 cols
  const int m = lane & 15, half = lane >> 4;
  const int nt0 = blockIdx.y * 8 + ng * 2;
  const int nt1 = nt0 + 1;
  const int nt0c = nt0 < NT ? nt0 : NT - 1; // clamp (compute redundant, store guarded)
  const int nt1c = nt1 < NT ? nt1 : NT - 1;

  const u16* b0p = wp + (long)nt0c * 512 + lane * 16;
  const u16* b1p = wp + (long)nt1c * 512 + lane * 16;
  const int r0 = (mg * 32 + m) * 32;        // LDS row base, m-tile 0
  const int r1 = (mg * 32 + 16 + m) * 32;   // LDS row base, m-tile 1
  const int kb = half * 16;

  v8f acc00 = {}, acc01 = {}, acc10 = {}, acc11 = {};
  for (int kc = 0; kc < KC; ++kc) {
    const u16* buf = lds + (kc & 1) * 2048;
    Frag a0, a1, b0, b1;
    a0.u[0] = *(const U128*)(buf + r0 + kb);
    a0.u[1] = *(const U128*)(buf + r0 + kb + 8);
    a1.u[0] = *(const U128*)(buf + r1 + kb);
    a1.u[1] = *(const U128*)(buf + r1 + kb + 8);
    b0.u[0] = *(const U128*)(b0p);
    b0.u[1] = *(const U128*)(b0p + 8);
    b1.u[0] = *(const U128*)(b1p);
    b1.u[1] = *(const U128*)(b1p + 8);
    __builtin_prefetch((const void*)(b0p + (long)NT * 512), 0, 1);
    if (kc + 1 < KC) stage(kc + 1, (kc + 1) & 1);
    acc00 = bwmma(a0, b0, acc00);
    acc01 = bwmma(a0, b1, acc01);
    acc10 = bwmma(a1, b0, acc10);
    acc11 = bwmma(a1, b1, acc11);
    b0p += (long)NT * 512;
    b1p += (long)NT * 512;
    async_wait0();
    __syncthreads();
  }

  const int n = lane & 15, mh = lane >> 4;
  auto store_tile = [&](const v8f& acc, int nt, long mbase) {
    if (nt >= NT) return;
    const int ngl = nt * 16 + n;
    float bb = (bias && ngl < Nact) ? bias[ngl] : 0.f;
    for (int r = 0; r < 8; ++r) {
      long p = mbase + mh * 8 + r;
      float v = acc[r] + bb;
      if (res) v += bf2f(res[p * (long)res_stride + ngl]);
      if (relu) v = fmaxf(v, 0.f);
      out[p * (long)out_stride + ngl] = f2bf(v);
    }
  };
  const long mb0 = p0 + mg * 32, mb1 = mb0 + 16;
  store_tile(acc00, nt0, mb0);
  store_tile(acc01, nt1, mb0);
  store_tile(acc10, nt0, mb1);
  store_tile(acc11, nt1, mb1);
}

// ---------------------------------------------------------------------------
// Build one shift's 416-wide MLP input rows (grid_sample nearest, zero pad).
// Row layout: [feat(128) | hr_guide(128) | hr_guide - lr_guide(128) | rel(2) | 0..]
// ---------------------------------------------------------------------------
__global__ void k_build_query(const float* __restrict__ coord,
                              const u16* __restrict__ feat,
                              const u16* __restrict__ hr_guide,
                              const u16* __restrict__ lr_guide,
                              u16* __restrict__ out, int shift) {
  const int n = blockIdx.x;
  const float vx = (shift & 2) ? 1.f : -1.f;
  const float vy = (shift & 1) ? 1.f : -1.f;
  const float cy = coord[n * 2 + 0], cx = coord[n * 2 + 1];
  const float sy = cy + vx / 32.f, sx = cx + vy / 32.f;
  // LR (32x32) nearest at shifted coord
  float fy = rintf((sy + 1.f) * 16.f - 0.5f);
  float fx = rintf((sx + 1.f) * 16.f - 0.5f);
  bool vlr = (fy >= 0.f) && (fy <= 31.f) && (fx >= 0.f) && (fx <= 31.f);
  int iy = min(max((int)fy, 0), 31), ix = min(max((int)fx, 0), 31);
  int plr = iy * 32 + ix;
  // HR (128x128) nearest at unshifted coord
  float hy = rintf((cy + 1.f) * 64.f - 0.5f);
  float hx = rintf((cx + 1.f) * 64.f - 0.5f);
  bool vhr = (hy >= 0.f) && (hy <= 127.f) && (hx >= 0.f) && (hx <= 127.f);
  int jy = min(max((int)hy, 0), 127), jx = min(max((int)hx, 0), 127);
  int phr = jy * 128 + jx;
  // q_coord from feature-grid coords (zero when sample invalid)
  float qcy = vlr ? (-1.f + 1.f / 32.f + (2.f / 32.f) * (float)iy) : 0.f;
  float qcx = vlr ? (-1.f + 1.f / 32.f + (2.f / 32.f) * (float)ix) : 0.f;
  float rely = (cy - qcy) * 32.f;
  float relx = (cx - qcx) * 32.f;

  u16* row = out + (long)n * 416;
  for (int c = threadIdx.x; c < 416; c += blockDim.x) {
    float v = 0.f;
    if (c < 128)      v = vlr ? bf2f(feat[(long)plr * 128 + c]) : 0.f;
    else if (c < 256) v = vhr ? bf2f(hr_guide[(long)phr * 128 + (c - 128)]) : 0.f;
    else if (c < 384) {
      float g = vhr ? bf2f(hr_guide[(long)phr * 128 + (c - 256)]) : 0.f;
      float l = vlr ? bf2f(lr_guide[(long)plr * 128 + (c - 256)]) : 0.f;
      v = g - l;
    } else if (c == 384) v = rely;
    else if (c == 385)   v = relx;
    row[c] = f2bf(v);
  }
}

// ---------------------------------------------------------------------------
// Softmax-blend 4 shifts, add lr_pixel, ReLU -> f32 output.
// ---------------------------------------------------------------------------
__global__ void k_combine(const u16* __restrict__ preds, const float* __restrict__ lr_pixel,
                          float* __restrict__ out, int Np) {
  int n = blockIdx.x * blockDim.x + threadIdx.x;
  if (n >= Np) return;
  float p0[4], p1[4];
  for (int s = 0; s < 4; ++s) {
    p0[s] = bf2f(preds[((long)s * Np + n) * 16 + 0]);
    p1[s] = bf2f(preds[((long)s * Np + n) * 16 + 1]);
  }
  float m = fmaxf(fmaxf(p1[0], p1[1]), fmaxf(p1[2], p1[3]));
  float e0 = __expf(p1[0] - m), e1 = __expf(p1[1] - m);
  float e2 = __expf(p1[2] - m), e3 = __expf(p1[3] - m);
  float sum = e0 + e1 + e2 + e3;
  float v = (p0[0] * e0 + p0[1] * e1 + p0[2] * e2 + p0[3] * e3) / sum;
  out[n] = fmaxf(lr_pixel[n] + v, 0.f);
}

// ---------------------------------------------------------------------------
// Host orchestration
// ---------------------------------------------------------------------------
extern "C" void kernel_launch(void* const* d_in, const int* in_sizes, int n_in,
                              void* d_out, int out_size, void* d_ws, size_t ws_size,
                              hipStream_t stream) {
  (void)in_sizes; (void)out_size; (void)ws_size;
  if (n_in < 115) return;
  char* ws = (char*)d_ws;
  size_t off = 0;
  auto alloc = [&](size_t elems) { size_t o = off; off += ((elems * 2 + 255) & ~(size_t)255); return o; };
  auto F = [&](int i) { return (const float*)d_in[i]; };
  auto P = [&](size_t o) { return (u16*)(ws + o); };
  auto pack = [&](int widx, size_t dst, int K, int N, int taps) {
    int Kpad = (K + 31) & ~31, Npad = (N + 15) & ~15;
    long total = (long)taps * Kpad * Npad;
    k_pack<<<(unsigned)((total + 255) / 256), 256, 0, stream>>>(
        F(widx), P(dst), K, N, Kpad / 32, Npad / 16, taps, total);
  };

  // --- pack all weights (pytree leaf order: alphabetical dict keys) ---
  struct RdnPk { size_t sfe1, sfe2, dense[4][4], lff[4], gff1, gff2, output; };
  RdnPk pkI, pkT;
  auto packRdn = [&](int pb, RdnPk& pk) {
    pk.sfe1 = alloc((size_t)9 * 32 * 128);  pack(pb + 46, pk.sfe1, 1, 128, 9);
    pk.sfe2 = alloc((size_t)9 * 128 * 128); pack(pb + 48, pk.sfe2, 128, 128, 9);
    for (int d = 0; d < 4; ++d) {
      for (int i = 0; i < 4; ++i) {
        int K = 128 * (i + 1);
        pk.dense[d][i] = alloc((size_t)9 * K * 128);
        pack(pb + 6 + d * 10 + i * 2, pk.dense[d][i], K, 128, 9);
      }
      pk.lff[d] = alloc((size_t)640 * 128);
      pack(pb + 6 + d * 10 + 8, pk.lff[d], 640, 128, 1);
    }
    pk.gff1 = alloc((size_t)512 * 128);       pack(pb + 0, pk.gff1, 512, 128, 1);
    pk.gff2 = alloc((size_t)9 * 128 * 128);   pack(pb + 2, pk.gff2, 128, 128, 9);
    pk.output = alloc((size_t)9 * 128 * 128); pack(pb + 4, pk.output, 128, 128, 9);
  };
  packRdn(5, pkI);   // int_params
  packRdn(55, pkT);  // tau_params
  size_t mlpw[5];
  {
    int Ks[5] = {386, 1024, 512, 256, 128};
    int Ns[5] = {1024, 512, 256, 128, 2};
    for (int i = 0; i < 5; ++i) {
      int Kp = (Ks[i] + 31) & ~31, Np = (Ns[i] + 15) & ~15;
      mlpw[i] = alloc((size_t)Kp * Np);
      pack(105 + i * 2, mlpw[i], Ks[i], Ns[i], 1);
    }
  }

  // --- activation buffers (bf16 [pixel][channel]) ---
  size_t actin  = alloc(16384);
  size_t sfe1b  = alloc((size_t)16384 * 128);
  size_t xb     = alloc((size_t)16384 * 128);
  size_t tmpb   = alloc((size_t)16384 * 128);
  size_t concat = alloc((size_t)16384 * 640);
  size_t locsb  = alloc((size_t)16384 * 512);
  size_t hrg    = alloc((size_t)16384 * 128);
  size_t lrg    = alloc((size_t)1024 * 128);
  size_t ftb    = alloc((size_t)1024 * 128);
  size_t qin    = alloc((size_t)16384 * 416);
  size_t h1     = alloc((size_t)16384 * 1024);
  size_t h2     = alloc((size_t)16384 * 512);
  size_t h3     = alloc((size_t)16384 * 256);
  size_t h4     = alloc((size_t)16384 * 128);
  size_t predb  = alloc((size_t)4 * 16384 * 16);

  auto conv3 = [&](const u16* in, int in_stride, int Cin, int CinPad, size_t w,
                   const float* bias, u16* out, int out_stride,
                   const u16* res, int res_stride, int H, int W, int relu) {
    dim3 g((unsigned)(H * W / 32));
    k_conv3x3<<<g, 256, 0, stream>>>(in, in_stride, Cin, CinPad, (const u16*)(ws + w),
                                     bias, out, out_stride, res, res_stride, H, W, relu);
  };
  auto gemm = [&](const u16* A, int M, int Kpad, size_t w, const float* bias,
                  int Nact, int Npad, u16* out, int out_stride,
                  const u16* res, int res_stride, int relu) {
    int NT = Npad / 16;
    dim3 g((unsigned)(M / 64), (unsigned)((NT + 7) / 8));
    k_gemm<<<g, 256, 0, stream>>>(A, Kpad, (const u16*)(ws + w), bias, Nact,
                                  out, out_stride, res, res_stride, NT, relu);
  };

  auto run_rdn = [&](const float* img, int H, int W, const RdnPk& pk, int pb, u16* guide) {
    int HWp = H * W;
    k_cvt<<<(HWp + 255) / 256, 256, 0, stream>>>(img, P(actin), HWp);
    conv3(P(actin), 1, 1, 32, pk.sfe1, F(pb + 47), P(sfe1b), 128, nullptr, 0, H, W, 0);
    conv3(P(sfe1b), 128, 128, 128, pk.sfe2, F(pb + 49), P(concat), 640, nullptr, 0, H, W, 0);
    for (int d = 0; d < 4; ++d) {
      for (int i = 0; i < 4; ++i) {
        int Cin = 128 * (i + 1);
        conv3(P(concat), 640, Cin, Cin, pk.dense[d][i], F(pb + 6 + d * 10 + i * 2 + 1),
              P(concat) + 128 + 128 * i, 640, nullptr, 0, H, W, 1);
      }
      // LFF 1x1 over 640ch + residual old-x (concat cols 0..127) -> locs slice d
      gemm(P(concat), HWp, 640, pk.lff[d], F(pb + 6 + d * 10 + 9), 128, 128,
           P(locsb) + d * 128, 512, P(concat), 640, 0);
      if (d < 3) {
        long tot = (long)HWp * 128;
        k_copy<<<(unsigned)((tot + 255) / 256), 256, 0, stream>>>(
            P(locsb) + d * 128, 512, P(concat), 640, HWp, 128);
      }
    }
    gemm(P(locsb), HWp, 512, pk.gff1, F(pb + 1), 128, 128, P(xb), 128, nullptr, 0, 0);
    conv3(P(xb), 128, 128, 128, pk.gff2, F(pb + 3), P(tmpb), 128, P(sfe1b), 128, H, W, 0);
    conv3(P(tmpb), 128, 128, 128, pk.output, F(pb + 5), guide, 128, nullptr, 0, H, W, 0);
  };

  run_rdn(F(0), 128, 128, pkI, 5, P(hrg));   // hr_guide = RDN(hr_int; int_params)
  run_rdn(F(4), 32, 32, pkI, 5, P(lrg));     // lr_guide = RDN(lr_int; int_params)
  run_rdn(F(1), 32, 32, pkT, 55, P(ftb));    // feat     = RDN(lr_tau; tau_params)

  for (int s = 0; s < 4; ++s) {
    k_build_query<<<16384, 128, 0, stream>>>(F(2), P(ftb), P(hrg), P(lrg), P(qin), s);
    gemm(P(qin), 16384, 416,  mlpw[0], F(106), 1024, 1024, P(h1), 1024, nullptr, 0, 1);
    gemm(P(h1),  16384, 1024, mlpw[1], F(108), 512,  512,  P(h2), 512,  nullptr, 0, 1);
    gemm(P(h2),  16384, 512,  mlpw[2], F(110), 256,  256,  P(h3), 256,  nullptr, 0, 1);
    gemm(P(h3),  16384, 256,  mlpw[3], F(112), 128,  128,  P(h4), 128,  nullptr, 0, 1);
    gemm(P(h4),  16384, 128,  mlpw[4], F(114), 2,    16,
         P(predb) + (size_t)s * 16384 * 16, 16, nullptr, 0, 0);
  }
  k_combine<<<64, 256, 0, stream>>>(P(predb), F(3), (float*)d_out, 16384);
}
